// VariationalRNN_44461501448838
// MI455X (gfx1250) — compile-verified, matrix operand order checked by reading, not run on hardware
//
#include <hip/hip_runtime.h>
#include <hip/hip_bf16.h>
#include <math.h>

// ---------------- types ----------------
typedef __attribute__((ext_vector_type(16))) __bf16 bf16x16;
typedef __attribute__((ext_vector_type(8)))  __bf16 bf16x8;
typedef __attribute__((ext_vector_type(8)))  float  v8f;

// global (addrspace 1) pointer types so weight loads lower to global_load_b128
typedef const __bf16  __attribute__((address_space(1)))* wgptr;
typedef const bf16x8  __attribute__((address_space(1)))* wgv8p;

#define Tn   512
#define Bn   1024
#define OBSn 64
#define Hn   128
#define Zn   64
#define EPSf 0.01f

#define STRA 264   // elem stride for 256-wide bf16 LDS buffers (528B: 16B aligned, bank-spread)
#define STRB 136   // elem stride for 128/64-wide bf16 LDS buffers (272B)
#define STRF 132   // elem stride for 128-wide f32 LDS buffers

// ws bf16 element offsets (fragment-layout weights)
enum : int {
  WS_PHIS = 0,                    // 128x64  -> 8*2  tiles
  WS_PHIZ = WS_PHIS + 8192,       // 128x64
  WS_P1   = WS_PHIZ + 8192,       // 128x128 -> 8*4
  WS_P2   = WS_P1   + 16384,      // 128x128
  WS_E1   = WS_P2   + 16384,      // 128x256 -> 8*8
  WS_E2   = WS_E1   + 32768,      // 128x128
  WS_IH   = WS_E2   + 16384,      // 512x256 -> 32*8
  WS_HH   = WS_IH   + 131072,     // 512x128 -> 32*4
  WS_D1   = WS_HH   + 65536,      // 128x256
  WS_D2   = WS_D1   + 32768,      // 64x128  -> 4*4
  WS_TOT  = WS_D2   + 8192
};

#define WMMA_BF16(a, b, c) \
  __builtin_amdgcn_wmma_f32_16x16x32_bf16(false, (a), false, (b), (short)0, (c), false, false)

// ---------------- weight conversion: fp32 [Out,In] row-major -> bf16 B-fragments ----------------
__global__ void vrnn_convert_w(const float* __restrict__ src, __bf16* __restrict__ dst,
                               int Out, int In) {
  int idx = blockIdx.x * blockDim.x + threadIdx.x;
  int Kt = In >> 5;
  int total = (Out >> 4) * Kt * 32;
  if (idx >= total) return;
  int lane = idx & 31;
  int tile = idx >> 5;            // tile = nt*Kt + kt
  int kt = tile % Kt;
  int nt = tile / Kt;
  int n  = nt * 16 + (lane & 15);
  int k0 = kt * 32 + ((lane >> 4) << 4);
  const float* sp = src + (size_t)n * In + k0;
  __bf16* dp = dst + (size_t)tile * 512 + lane * 16;
#pragma unroll
  for (int i = 0; i < 16; ++i) dp[i] = (__bf16)sp[i];
}

__global__ void vrnn_zero_out(float* out) { out[0] = 0.f; }

// ---------------- fragment helpers ----------------
__device__ __forceinline__ bf16x16 ldsA(const __bf16* buf, int stride, int kbase, int lane) {
  int m = lane & 15;
  int k = kbase + ((lane >> 4) << 4);
  union { bf16x16 v; bf16x8 h[2]; } u;
  const bf16x8* p = (const bf16x8*)(buf + m * stride + k);
  u.h[0] = p[0]; u.h[1] = p[1];
  return u.v;
}
__device__ __forceinline__ bf16x16 ldB(wgptr w, int tile, int lane) {
  union { bf16x16 v; bf16x8 h[2]; } u;
  wgv8p p = (wgv8p)(w + (size_t)tile * 512 + lane * 16);
  u.h[0] = p[0]; u.h[1] = p[1];
  return u.v;
}
__device__ __forceinline__ v8f splat8(float v) {
  v8f a;
#pragma unroll
  for (int r = 0; r < 8; ++r) a[r] = v;
  return a;
}
__device__ __forceinline__ void stC_bf16(__bf16* buf, int stride, int colbase, int lane,
                                         v8f a, bool relu) {
  int m0 = (lane >> 4) * 8;
  int col = colbase + (lane & 15);
#pragma unroll
  for (int r = 0; r < 8; ++r) {
    float v = a[r];
    if (relu) v = v > 0.f ? v : 0.f;
    buf[(m0 + r) * stride + col] = (__bf16)v;
  }
}
__device__ __forceinline__ void stC_f32(float* buf, int stride, int colbase, int lane, v8f a) {
  int m0 = (lane >> 4) * 8;
  int col = colbase + (lane & 15);
#pragma unroll
  for (int r = 0; r < 8; ++r) buf[(m0 + r) * stride + col] = a[r];
}
__device__ __forceinline__ float softplusf(float x) {
  return (x > 20.f) ? x : log1pf(expf(x));
}
__device__ __forceinline__ float sigmoidf_(float x) { return 1.f / (1.f + expf(-x)); }

// ---------------- persistent recurrence kernel: 1 block = 16 batch rows ----------------
__global__ __launch_bounds__(256, 1) void vrnn_main(
    const float* __restrict__ s, const float* __restrict__ noise,
    const float* __restrict__ b_phi_s, const float* __restrict__ b_phi_z,
    const float* __restrict__ bp1, const float* __restrict__ bp2,
    const float* __restrict__ be1, const float* __restrict__ be2,
    const float* __restrict__ b_ih, const float* __restrict__ b_hh,
    const float* __restrict__ bd1, const float* __restrict__ bd2,
    const __bf16* __restrict__ ws, float* __restrict__ out) {
  __shared__ __bf16 xe[16 * STRA];   // [phi_s | h]  (256 cols)
  __shared__ __bf16 xd[16 * STRA];   // [phi_z | h]  (256 cols)
  __shared__ __bf16 hid[16 * STRB];  // hidden scratch (128 cols), reused
  __shared__ __bf16 zb[16 * STRB];   // z_samp bf16 (64 cols)
  __shared__ __bf16 sb[16 * STRB];   // s_t bf16 (64 cols)
  __shared__ float  sf[16 * 68];     // s_t f32
  __shared__ float  eo[16 * STRF];   // encoder output f32 (128 cols)
  __shared__ float  po[16 * STRF];   // prior output f32
  __shared__ float  red[8];

  const int tid  = threadIdx.x;
  const int wave = tid >> 5;
  const int lane = tid & 31;
  const int cc   = lane & 15;
  const int b0   = blockIdx.x * 16;

  for (int i = tid; i < 16 * STRA; i += 256) { xe[i] = (__bf16)0.f; xd[i] = (__bf16)0.f; }

  // ---- preload all loop-invariant bias scalars into registers (12 VGPRs/lane) ----
  const float bPS = b_phi_s[wave * 16 + cc];
  const float bE1 = be1[wave * 16 + cc];
  const float bE2 = be2[wave * 16 + cc];
  const float bP1 = bp1[wave * 16 + cc];
  const float bP2 = bp2[wave * 16 + cc];
  const float bPZ = b_phi_z[wave * 16 + cc];
  const float bD1 = bd1[wave * 16 + cc];
  const float bD2 = (wave < 4) ? bd2[wave * 16 + cc] : 0.f;
  float bG[4];
#pragma unroll
  for (int gi = 0; gi < 4; ++gi) {
    int nt = gi * 8 + wave;
    bG[gi] = b_ih[nt * 16 + cc] + b_hh[nt * 16 + cc];
  }

  v8f cst;
#pragma unroll
  for (int r = 0; r < 8; ++r) cst[r] = 0.f;
  float loss = 0.f;
  __syncthreads();

  for (int t = 0; t < Tn; ++t) {
    // Opaque copy of the weight base pointer (explicit global addrspace):
    // defeats LICM (no hoist/spill of 84 B-tiles) while keeping the loads as
    // saddr-form global_load_b128 tracked only by LOADcnt.
    wgptr wv = (wgptr)ws;
    asm volatile("" : "+s"(wv));

    // ---- stage s_t tile (f32 + bf16) ----
    const float* sp = s + ((size_t)t * Bn + b0) * OBSn;
    for (int i = tid; i < 16 * OBSn; i += 256) {
      int m = i >> 6, c = i & 63;
      float v = sp[m * OBSn + c];
      sf[m * 68 + c]   = v;
      sb[m * STRB + c] = (__bf16)v;
    }
    if (t + 1 < Tn) __builtin_prefetch(sp + (size_t)Bn * OBSn + tid * 4, 0, 1);
    __syncthreads();

    // ---- phi_s = s_t @ W_phi_s^T + b  -> xe[:,0:128] ----
    {
      v8f acc = splat8(bPS);
#pragma unroll
      for (int kt = 0; kt < 2; ++kt)
        acc = WMMA_BF16(ldsA(sb, STRB, kt * 32, lane), ldB(wv + WS_PHIS, wave * 2 + kt, lane), acc);
      stC_bf16(xe, STRA, wave * 16, lane, acc, false);
    }
    __syncthreads();

    // ---- enc hidden = relu([phi_s,h] @ We1^T + be1) -> hid ----
    {
      v8f acc = splat8(bE1);
#pragma unroll
      for (int kt = 0; kt < 8; ++kt)
        acc = WMMA_BF16(ldsA(xe, STRA, kt * 32, lane), ldB(wv + WS_E1, wave * 8 + kt, lane), acc);
      stC_bf16(hid, STRB, wave * 16, lane, acc, true);
    }
    __syncthreads();

    // ---- enc out = hid @ We2^T + be2 -> eo (f32) ----
    {
      v8f acc = splat8(bE2);
#pragma unroll
      for (int kt = 0; kt < 4; ++kt)
        acc = WMMA_BF16(ldsA(hid, STRB, kt * 32, lane), ldB(wv + WS_E2, wave * 4 + kt, lane), acc);
      stC_f32(eo, STRF, wave * 16, lane, acc);
    }
    __syncthreads();

    // ---- prior hidden = relu(h @ Wp1^T + bp1) -> hid ----
    {
      v8f acc = splat8(bP1);
#pragma unroll
      for (int kt = 0; kt < 4; ++kt)
        acc = WMMA_BF16(ldsA(xe, STRA, 128 + kt * 32, lane), ldB(wv + WS_P1, wave * 4 + kt, lane), acc);
      stC_bf16(hid, STRB, wave * 16, lane, acc, true);
    }
    __syncthreads();

    // ---- prior out = hid @ Wp2^T + bp2 -> po (f32) ----
    {
      v8f acc = splat8(bP2);
#pragma unroll
      for (int kt = 0; kt < 4; ++kt)
        acc = WMMA_BF16(ldsA(hid, STRB, kt * 32, lane), ldB(wv + WS_P2, wave * 4 + kt, lane), acc);
      stC_f32(po, STRF, wave * 16, lane, acc);
    }
    __syncthreads();

    // ---- reparameterize + KL (elementwise over 16x64) ----
    {
      const float* np_ = noise + ((size_t)t * Bn + b0) * Zn;
      for (int i = tid; i < 16 * Zn; i += 256) {
        int m = i >> 6, j = i & 63;
        float zmu  = eo[m * STRF + j];
        float zstd = softplusf(eo[m * STRF + 64 + j]);
        float pmu  = po[m * STRF + j];
        float pstd = softplusf(po[m * STRF + 64 + j]);
        float zs   = zmu + zstd * np_[m * Zn + j];
        zb[m * STRB + j] = (__bf16)zs;
        float dm = zmu - pmu;
        float kl = 0.5f * (2.f * logf(pstd + EPSf) - 2.f * logf(zstd + EPSf)
                           + (zstd * zstd + dm * dm) / (expf(2.f * pstd) - 1.f));
        loss += kl * (1.f / (float)Bn);
      }
    }
    __syncthreads();

    // ---- phi_z = z_samp @ W_phi_z^T + b -> xd[:,0:128] ----
    {
      v8f acc = splat8(bPZ);
#pragma unroll
      for (int kt = 0; kt < 2; ++kt)
        acc = WMMA_BF16(ldsA(zb, STRB, kt * 32, lane), ldB(wv + WS_PHIZ, wave * 2 + kt, lane), acc);
      stC_bf16(xd, STRA, wave * 16, lane, acc, false);
    }
    __syncthreads();

    // ---- dec hidden = relu([phi_z,h] @ Wd1^T + bd1) -> hid ----
    {
      v8f acc = splat8(bD1);
#pragma unroll
      for (int kt = 0; kt < 8; ++kt)
        acc = WMMA_BF16(ldsA(xd, STRA, kt * 32, lane), ldB(wv + WS_D1, wave * 8 + kt, lane), acc);
      stC_bf16(hid, STRB, wave * 16, lane, acc, true);
    }
    __syncthreads();

    // ---- recon (waves 0..3, 64 output cols) + rc loss in registers ----
    if (wave < 4) {
      v8f acc = splat8(bD2);
#pragma unroll
      for (int kt = 0; kt < 4; ++kt)
        acc = WMMA_BF16(ldsA(hid, STRB, kt * 32, lane), ldB(wv + WS_D2, wave * 4 + kt, lane), acc);
      int m0 = (lane >> 4) * 8;
      int col = wave * 16 + cc;
#pragma unroll
      for (int r = 0; r < 8; ++r) {
        float d = acc[r] - sf[(m0 + r) * 68 + col];
        loss += 0.5f * d * d * (1.f / (float)Bn);
      }
    }

    // ---- LSTM gates: wave w owns i/f/g/o col-tile w -> elementwise entirely in registers ----
    v8f g[4];
#pragma unroll
    for (int gi = 0; gi < 4; ++gi) {
      int nt = gi * 8 + wave;
      v8f acc = splat8(bG[gi]);
#pragma unroll
      for (int kt = 0; kt < 8; ++kt) {
        bf16x16 a = (kt < 4) ? ldsA(xe, STRA, kt * 32, lane)
                             : ldsA(xd, STRA, (kt - 4) * 32, lane);
        acc = WMMA_BF16(a, ldB(wv + WS_IH, nt * 8 + kt, lane), acc);
      }
#pragma unroll
      for (int kt = 0; kt < 4; ++kt)
        acc = WMMA_BF16(ldsA(xe, STRA, 128 + kt * 32, lane),
                        ldB(wv + WS_HH, nt * 4 + kt, lane), acc);
      g[gi] = acc;
    }
    __syncthreads();  // all reads of old h complete before overwrite
    {
      int m0 = (lane >> 4) * 8;
      int col = wave * 16 + cc;
#pragma unroll
      for (int r = 0; r < 8; ++r) {
        float ig = sigmoidf_(g[0][r]);
        float fg = sigmoidf_(g[1][r]);
        float gg = tanhf(g[2][r]);
        float og = sigmoidf_(g[3][r]);
        float cn = fg * cst[r] + ig * gg;
        cst[r] = cn;
        __bf16 hb = (__bf16)(og * tanhf(cn));
        xe[(m0 + r) * STRA + 128 + col] = hb;
        xd[(m0 + r) * STRA + 128 + col] = hb;
      }
    }
    __syncthreads();
  }

  // ---- block reduce + single atomic ----
#pragma unroll
  for (int off = 16; off > 0; off >>= 1) loss += __shfl_down(loss, off, 32);
  if (lane == 0) red[wave] = loss;
  __syncthreads();
  if (wave == 0) {
    float v = (lane < 8) ? red[lane] : 0.f;
#pragma unroll
    for (int off = 4; off > 0; off >>= 1) v += __shfl_down(v, off, 32);
    if (lane == 0) atomicAdd(out, v);
  }
}

// ---------------- launch ----------------
extern "C" void kernel_launch(void* const* d_in, const int* in_sizes, int n_in,
                              void* d_out, int out_size, void* d_ws, size_t ws_size,
                              hipStream_t stream) {
  const float* s       = (const float*)d_in[0];
  const float* noise   = (const float*)d_in[1];
  const float* W_phi_s = (const float*)d_in[2];
  const float* b_phi_s = (const float*)d_in[3];
  const float* W_phi_z = (const float*)d_in[4];
  const float* b_phi_z = (const float*)d_in[5];
  const float* Wp1     = (const float*)d_in[6];
  const float* bp1     = (const float*)d_in[7];
  const float* Wp2     = (const float*)d_in[8];
  const float* bp2     = (const float*)d_in[9];
  const float* We1     = (const float*)d_in[10];
  const float* be1     = (const float*)d_in[11];
  const float* We2     = (const float*)d_in[12];
  const float* be2     = (const float*)d_in[13];
  const float* W_ih    = (const float*)d_in[14];
  const float* b_ih    = (const float*)d_in[15];
  const float* W_hh    = (const float*)d_in[16];
  const float* b_hh    = (const float*)d_in[17];
  const float* Wd1     = (const float*)d_in[18];
  const float* bd1     = (const float*)d_in[19];
  const float* Wd2     = (const float*)d_in[20];
  const float* bd2     = (const float*)d_in[21];

  __bf16* ws = (__bf16*)d_ws;
  auto conv = [&](const float* W, int off, int Out, int In) {
    int total = (Out / 16) * (In / 32) * 32;
    vrnn_convert_w<<<(total + 255) / 256, 256, 0, stream>>>(W, ws + off, Out, In);
  };
  conv(W_phi_s, WS_PHIS, 128, 64);
  conv(W_phi_z, WS_PHIZ, 128, 64);
  conv(Wp1,     WS_P1,   128, 128);
  conv(Wp2,     WS_P2,   128, 128);
  conv(We1,     WS_E1,   128, 256);
  conv(We2,     WS_E2,   128, 128);
  conv(W_ih,    WS_IH,   512, 256);
  conv(W_hh,    WS_HH,   512, 128);
  conv(Wd1,     WS_D1,   128, 256);
  conv(Wd2,     WS_D2,   64,  128);

  vrnn_zero_out<<<1, 1, 0, stream>>>((float*)d_out);

  vrnn_main<<<Bn / 16, 256, 0, stream>>>(
      s, noise, b_phi_s, b_phi_z, bp1, bp2, be1, be2,
      b_ih, b_hh, bd1, bd2, ws, (float*)d_out);
}
